// HRRAttention_17540646437771
// MI455X (gfx1250) — matrix operand order repaired and checked by reading, not compiled, not run on hardware
//
#include <hip/hip_runtime.h>

typedef __attribute__((ext_vector_type(16))) __bf16 v16bf;
typedef __attribute__((ext_vector_type(8)))  float  v8f;

#define B_  4
#define S_  2048
#define D_  1024
#define H_  16
#define HD_ 64
#define M_  (B_ * S_)   // 8192

// ---------------------------------------------------------------------------
// fp32 -> bf16 hi/lo split helpers (stored as ushort to avoid __bf16 scalar ops)
// ---------------------------------------------------------------------------
static __device__ __forceinline__ unsigned short f32_to_bf16_rne(float f) {
    unsigned int u = __float_as_uint(f);
    unsigned int lsb = (u >> 16) & 1u;
    u += 0x7fffu + lsb;
    return (unsigned short)(u >> 16);
}
static __device__ __forceinline__ float bf16_to_f32(unsigned short h) {
    return __uint_as_float(((unsigned int)h) << 16);
}

__global__ void split_kernel(const float* __restrict__ src,
                             unsigned short* __restrict__ hi,
                             unsigned short* __restrict__ lo, long n) {
    long i = (long)blockIdx.x * blockDim.x + threadIdx.x;
    if (i >= n) return;
    float x = src[i];
    unsigned short h = f32_to_bf16_rne(x);
    hi[i] = h;
    lo[i] = f32_to_bf16_rne(x - bf16_to_f32(h));
}

// A[m,k] = w[b,h,s] * vp[m,k], split to bf16 hi/lo  (m = b*S+s, h = k>>6)
__global__ void scale_split_kernel(const float* __restrict__ vp,
                                   const float* __restrict__ w,
                                   unsigned short* __restrict__ hi,
                                   unsigned short* __restrict__ lo, long n) {
    long i = (long)blockIdx.x * blockDim.x + threadIdx.x;
    if (i >= n) return;
    int  kcol = (int)(i & (D_ - 1));
    long m    = i >> 10;                 // D_ == 1024
    int  s    = (int)(m & (S_ - 1));
    int  b    = (int)(m >> 11);         // S_ == 2048
    int  h    = kcol >> 6;
    float x = vp[i] * w[((size_t)b * H_ + h) * S_ + s];
    unsigned short hh = f32_to_bf16_rne(x);
    hi[i] = hh;
    lo[i] = f32_to_bf16_rne(x - bf16_to_f32(hh));
}

__global__ void zero_kernel(float* __restrict__ p, int n) {
    int i = blockIdx.x * blockDim.x + threadIdx.x;
    if (i < n) p[i] = 0.f;
}

// ---------------------------------------------------------------------------
// GEMM: C[M,N] = A[M,K] @ B[N,K]^T + bias[N], bf16x3 (hi*hi + hi*lo + lo*hi)
// block = 256 threads = 8 waves (wave32). Wave tile 32x64, block tile 128x128.
// ---------------------------------------------------------------------------
__global__ __launch_bounds__(256) void gemm_bf16x3_kernel(
    const unsigned short* __restrict__ Ahi, const unsigned short* __restrict__ Alo,
    const unsigned short* __restrict__ Bhi, const unsigned short* __restrict__ Blo,
    const float* __restrict__ bias, float* __restrict__ C,
    int M, int N, int K)
{
    const int lane  = threadIdx.x & 31;
    const int wave  = threadIdx.x >> 5;   // 0..7
    const int waveM = wave & 3;           // 4 waves along M
    const int waveN = wave >> 2;          // 2 waves along N
    const int m0 = blockIdx.y * 128 + waveM * 32;
    const int n0 = blockIdx.x * 128 + waveN * 64;
    const int lhalf = lane >> 4;          // 0|1
    const int l16   = lane & 15;

    v8f acc[2][4];
#pragma unroll
    for (int i = 0; i < 2; ++i)
#pragma unroll
        for (int j = 0; j < 4; ++j) acc[i][j] = (v8f){};

    union V16 { v16bf v; uint4 u[2]; };

    for (int k0 = 0; k0 < K; k0 += 32) {
        V16 ahi[2], alo[2], bhi[4], blo[4];
        // A tile (16x32 bf16): lane<16 -> K 0..7 & 16..23 ; lane>=16 -> K 8..15 & 24..31
#pragma unroll
        for (int mt = 0; mt < 2; ++mt) {
            const size_t base = (size_t)(m0 + mt * 16 + l16) * K + k0 + lhalf * 8;
            ahi[mt].u[0] = *(const uint4*)(Ahi + base);
            ahi[mt].u[1] = *(const uint4*)(Ahi + base + 16);
            alo[mt].u[0] = *(const uint4*)(Alo + base);
            alo[mt].u[1] = *(const uint4*)(Alo + base + 16);
        }
        // B tile (32x16 bf16): lane&15 = N col; lane<16 -> K 0..7 & 16..23 ; lane>=16 -> K 8..15 & 24..31
#pragma unroll
        for (int nt = 0; nt < 4; ++nt) {
            const size_t base = (size_t)(n0 + nt * 16 + l16) * K + k0 + lhalf * 8;
            bhi[nt].u[0] = *(const uint4*)(Bhi + base);
            bhi[nt].u[1] = *(const uint4*)(Bhi + base + 16);
            blo[nt].u[0] = *(const uint4*)(Blo + base);
            blo[nt].u[1] = *(const uint4*)(Blo + base + 16);
        }
        if (k0 + 32 < K) {
            __builtin_prefetch(Ahi + (size_t)(m0 + l16) * K + k0 + 32, 0, 0);
            __builtin_prefetch(Bhi + (size_t)(n0 + l16) * K + k0 + 32, 0, 0);
        }
#pragma unroll
        for (int mt = 0; mt < 2; ++mt)
#pragma unroll
            for (int nt = 0; nt < 4; ++nt) {
                acc[mt][nt] = __builtin_amdgcn_wmma_f32_16x16x32_bf16(
                    false, ahi[mt].v, false, bhi[nt].v, (short)0, acc[mt][nt], false, false);
                acc[mt][nt] = __builtin_amdgcn_wmma_f32_16x16x32_bf16(
                    false, ahi[mt].v, false, blo[nt].v, (short)0, acc[mt][nt], false, false);
                acc[mt][nt] = __builtin_amdgcn_wmma_f32_16x16x32_bf16(
                    false, alo[mt].v, false, bhi[nt].v, (short)0, acc[mt][nt], false, false);
            }
    }

    // Epilogue: C/D layout -> N = l16, M = r + 8*lhalf within each 16x16 tile
#pragma unroll
    for (int nt = 0; nt < 4; ++nt) {
        const int n  = n0 + nt * 16 + l16;
        const float bn = bias[n];
#pragma unroll
        for (int mt = 0; mt < 2; ++mt) {
#pragma unroll
            for (int r = 0; r < 8; ++r) {
                const int m = m0 + mt * 16 + lhalf * 8 + r;
                C[(size_t)m * N + n] = acc[mt][nt][r] + bn;
            }
        }
    }
}

// ---------------------------------------------------------------------------
// bind: beta[b,h,j] = sum_s sum_i kp[b,s,h,i] * vp[b,s,h,(j-i)&63]
// grid (B*H, 16 s-chunks), block 64; atomic accumulation into beta
// ---------------------------------------------------------------------------
__global__ __launch_bounds__(64) void bind_kernel(
    const float* __restrict__ kp, const float* __restrict__ vp,
    float* __restrict__ beta)
{
    const int bh = blockIdx.x;          // b*H + h
    const int b  = bh >> 4, h = bh & 15;
    const int j  = threadIdx.x;
    const int sPer = S_ / gridDim.y;
    const int sBeg = blockIdx.y * sPer;
    __shared__ float kr[64], vr[64];
    const size_t rowStride = (size_t)H_ * HD_;          // 1024
    const float* kbase = kp + (size_t)b * S_ * rowStride + (size_t)h * HD_;
    const float* vbase = vp + (size_t)b * S_ * rowStride + (size_t)h * HD_;
    float acc = 0.f;
    for (int s = sBeg; s < sBeg + sPer; ++s) {
        kr[j] = kbase[(size_t)s * rowStride + j];
        vr[j] = vbase[(size_t)s * rowStride + j];
        __syncthreads();
#pragma unroll
        for (int i = 0; i < 64; ++i) acc += kr[i] * vr[(j - i) & 63];
        __syncthreads();
    }
    atomicAdd(&beta[(size_t)bh * 64 + j], acc);
}

// ---------------------------------------------------------------------------
// unbind + cosine: v_hat[j] = sum_i beta[i]*qp[(i-j)&63]; a[b,h,s] = cos(vp,v_hat)
// block 256 = 4 sub-groups of 64, one (b,s,h) triple each
// ---------------------------------------------------------------------------
__global__ __launch_bounds__(256) void unbind_cos_kernel(
    const float* __restrict__ qp, const float* __restrict__ vp,
    const float* __restrict__ beta, float* __restrict__ a)
{
    const int sub = threadIdx.x >> 6;   // 0..3
    const int j   = threadIdx.x & 63;
    const long t  = (long)blockIdx.x * 4 + sub;   // index over [B,S,H]
    const int h   = (int)(t & (H_ - 1));
    const long bs = t >> 4;
    const int s   = (int)(bs & (S_ - 1));
    const int b   = (int)(bs >> 11);
    __shared__ float q_sh[4][64];
    __shared__ float bet_sh[4][64];
    __shared__ float red[3][256];
    const float* qrow = qp + (size_t)t * 64;
    const float* vrow = vp + (size_t)t * 64;
    q_sh[sub][j]   = qrow[j];
    bet_sh[sub][j] = beta[((size_t)b * H_ + h) * 64 + j];
    const float vj = vrow[j];
    __syncthreads();
    float vh = 0.f;
#pragma unroll
    for (int i = 0; i < 64; ++i) vh += bet_sh[sub][i] * q_sh[sub][(i - j) & 63];
    red[0][threadIdx.x] = vj * vh;
    red[1][threadIdx.x] = vj * vj;
    red[2][threadIdx.x] = vh * vh;
    __syncthreads();
    for (int st = 32; st > 0; st >>= 1) {
        if (j < st) {
            red[0][threadIdx.x] += red[0][threadIdx.x + st];
            red[1][threadIdx.x] += red[1][threadIdx.x + st];
            red[2][threadIdx.x] += red[2][threadIdx.x + st];
        }
        __syncthreads();
    }
    if (j == 0) {
        float dot = red[0][sub * 64];
        float n1  = fmaxf(sqrtf(red[1][sub * 64]), 1e-8f);
        float n2  = fmaxf(sqrtf(red[2][sub * 64]), 1e-8f);
        a[((size_t)b * H_ + h) * S_ + s] = dot / (n1 * n2);
    }
}

// softmax over S, in place on a[B*H, S]
__global__ __launch_bounds__(256) void softmax_kernel(float* __restrict__ a)
{
    float* row = a + (size_t)blockIdx.x * S_;
    __shared__ float red[256];
    float m = -INFINITY;
    for (int s = threadIdx.x; s < S_; s += 256) m = fmaxf(m, row[s]);
    red[threadIdx.x] = m; __syncthreads();
    for (int st = 128; st > 0; st >>= 1) {
        if (threadIdx.x < st) red[threadIdx.x] = fmaxf(red[threadIdx.x], red[threadIdx.x + st]);
        __syncthreads();
    }
    m = red[0]; __syncthreads();
    float sum = 0.f;
    for (int s = threadIdx.x; s < S_; s += 256) { float e = expf(row[s] - m); row[s] = e; sum += e; }
    red[threadIdx.x] = sum; __syncthreads();
    for (int st = 128; st > 0; st >>= 1) {
        if (threadIdx.x < st) red[threadIdx.x] += red[threadIdx.x + st];
        __syncthreads();
    }
    const float inv = 1.f / red[0];
    __syncthreads();
    for (int s = threadIdx.x; s < S_; s += 256) row[s] *= inv;
}

// ---------------------------------------------------------------------------
extern "C" void kernel_launch(void* const* d_in, const int* in_sizes, int n_in,
                              void* d_out, int out_size, void* d_ws, size_t ws_size,
                              hipStream_t stream)
{
    const float* q  = (const float*)d_in[0];
    const float* k  = (const float*)d_in[1];
    const float* v  = (const float*)d_in[2];
    const float* Wq = (const float*)d_in[3];
    const float* bq = (const float*)d_in[4];
    const float* Wk = (const float*)d_in[5];
    const float* bk = (const float*)d_in[6];
    const float* Wv = (const float*)d_in[7];
    const float* bv = (const float*)d_in[8];
    const float* Wo = (const float*)d_in[9];
    const float* bo = (const float*)d_in[10];
    float* out = (float*)d_out;

    const long MK = (long)M_ * D_;        // 8388608
    const long NK = (long)D_ * D_;        // 1048576

    char* ws = (char*)d_ws;
    size_t off = 0;
    auto alloc = [&](size_t bytes) { char* p = ws + off; off += (bytes + 255) & ~(size_t)255; return p; };
    float*          qp   = (float*)         alloc(MK * 4);
    float*          kp   = (float*)         alloc(MK * 4);
    float*          vp   = (float*)         alloc(MK * 4);
    unsigned short* Ahi  = (unsigned short*)alloc(MK * 2);
    unsigned short* Alo  = (unsigned short*)alloc(MK * 2);
    unsigned short* Whi  = (unsigned short*)alloc(NK * 2);
    unsigned short* Wlo  = (unsigned short*)alloc(NK * 2);
    float*          beta = (float*)         alloc((size_t)B_ * H_ * HD_ * 4);
    float*          aw   = (float*)         alloc((size_t)B_ * H_ * S_ * 4);

    const dim3 gemmGrid(D_ / 128, M_ / 128);  // (8, 64)

    auto proj = [&](const float* X, const float* W, const float* bias, float* P) {
        split_kernel<<<(int)(MK / 256), 256, 0, stream>>>(X, Ahi, Alo, MK);
        split_kernel<<<(int)(NK / 256), 256, 0, stream>>>(W, Whi, Wlo, NK);
        gemm_bf16x3_kernel<<<gemmGrid, 256, 0, stream>>>(Ahi, Alo, Whi, Wlo, bias, P, M_, D_, D_);
    };

    proj(q, Wq, bq, qp);
    proj(k, Wk, bk, kp);
    proj(v, Wv, bv, vp);

    zero_kernel<<<(B_ * H_ * HD_ + 255) / 256, 256, 0, stream>>>(beta, B_ * H_ * HD_);
    bind_kernel<<<dim3(B_ * H_, 16), 64, 0, stream>>>(kp, vp, beta);
    unbind_cos_kernel<<<(B_ * S_ * H_) / 4, 256, 0, stream>>>(qp, vp, beta, aw);
    softmax_kernel<<<B_ * H_, 256, 0, stream>>>(aw);

    scale_split_kernel<<<(int)(MK / 256), 256, 0, stream>>>(vp, aw, Ahi, Alo, MK);
    split_kernel<<<(int)(NK / 256), 256, 0, stream>>>(Wo, Whi, Wlo, NK);
    gemm_bf16x3_kernel<<<gemmGrid, 256, 0, stream>>>(Ahi, Alo, Whi, Wlo, bo, out, M_, D_, D_);
}
